// ASFGW_Layer_60662118089058
// MI455X (gfx1250) — matrix operand order repaired and checked by reading, not compiled, not run on hardware
//
#include <hip/hip_runtime.h>
#include <hip/hip_bf16.h>
#include <math.h>

typedef _Float16 h16;
typedef __attribute__((ext_vector_type(16))) _Float16 v16h;
typedef __attribute__((ext_vector_type(8)))  float    v8f;

#define B_    8192
#define M_    10
#define NN_   9
#define FIN_  128
#define DX_   128
#define K_    64
#define L_    32
#define NALL_ 100000
#define BM_   (B_*M_)              /* 81920 */
#define PR_OFF BM_
#define PN_OFF (BM_ + K_)
#define RTOT_ (BM_ + K_ + K_*NN_)  /* 82560, multiple of 16 */
#define CSTR_ (2*NN_*L_)           /* 576 = 18*32 */

/* ---------------- 9-element sorting network (25 CEs, constant indices) ------- */
#define SORT9_NET(OP) \
  OP(0,3) OP(1,7) OP(2,5) OP(4,8) \
  OP(0,7) OP(2,4) OP(3,8) OP(5,6) \
  OP(0,2) OP(1,3) OP(4,5) OP(7,8) \
  OP(1,4) OP(3,6) OP(5,7) \
  OP(0,1) OP(2,4) OP(3,5) OP(6,8) \
  OP(2,3) OP(4,5) OP(6,7) \
  OP(1,2) OP(3,4) OP(5,6)

__device__ __forceinline__ void sort9(float* v) {
#define CSW_(i,j) { if (v[i] > v[j]) { float t=v[i]; v[i]=v[j]; v[j]=t; } }
  SORT9_NET(CSW_)
#undef CSW_
}
__device__ __forceinline__ void sort9p(float* v, float* w) {
#define CSWP_(i,j) { if (v[i] > v[j]) { float t=v[i]; v[i]=v[j]; v[j]=t; \
                                        t=w[i]; w[i]=w[j]; w[j]=t; } }
  SORT9_NET(CSWP_)
#undef CSWP_
}

/* ---------------- WMMA helpers (CDNA5 16x16x32 f16 -> f32) ------------------- */
__device__ __forceinline__ v8f wmma16(v16h a, v16h b, v8f c) {
  return __builtin_amdgcn_wmma_f32_16x16x32_f16(false, a, false, b, (short)0, c,
                                                false, false);
}

/* A-matrix 16x32 f16 fragment: lane m = lane&15, half h = lane>>4.
   Two contiguous 16B runs -> compiler emits b128 loads. */
__device__ __forceinline__ v16h frag_a_f16(const h16* p, int lane_hi8) {
  v16h a;
#pragma unroll
  for (int v = 0; v < 8; ++v) a[v]   = p[2*lane_hi8 + v];
#pragma unroll
  for (int v = 0; v < 8; ++v) a[8+v] = p[2*lane_hi8 + 8 + v];
  return a;
}
__device__ __forceinline__ v16h frag_a_f32cvt(const float* p, int lane_hi8) {
  v16h a;
#pragma unroll
  for (int v = 0; v < 8; ++v) a[v]   = (h16)p[2*lane_hi8 + v];
#pragma unroll
  for (int v = 0; v < 8; ++v) a[8+v] = (h16)p[2*lane_hi8 + 8 + v];
  return a;
}

/* B fragments are stored PRE-SHUFFLED: [ktile][ntile][lane][16 f16 contiguous],
   so one fragment = one aligned 32-byte v16h load.
   Logical element (row=k, col=n) of the B matrix maps to:
     ktile = row>>5, half = (row>>4)&1, i = row&15,
     ntile = col>>4, lane = half*16 + (col&15),
     index = ((ktile*NT + ntile)*32 + lane)*16 + i                          */
__device__ __forceinline__ int remapB(int row, int col, int ntiles) {
  return ((((row >> 5) * ntiles + (col >> 4)) * 32)
          + (((row >> 4) & 1) * 16) + (col & 15)) * 16 + (row & 15);
}

/* ============ k_prep: normalized projection directions ====================== */
__global__ void k_prep(const float* theta_x, const float* theta_s,
                       float* tnx, float* tns) {
  int l = threadIdx.x;
  if (l < L_) {
    float s = 0.f;
    for (int d = 0; d < DX_; ++d) { float v = theta_x[l*DX_+d]; s += v*v; }
    float inv = rsqrtf(s);
    for (int d = 0; d < DX_; ++d) tnx[l*DX_+d] = theta_x[l*DX_+d]*inv;
    float s2 = 0.f;
    for (int d = 0; d < NN_; ++d) { float v = theta_s[l*NN_+d]; s2 += v*v; }
    float inv2 = rsqrtf(s2);
    for (int d = 0; d < NN_; ++d) tns[l*NN_+d] = theta_s[l*NN_+d]*inv2;
  }
}

/* ============ k_lin_ln: gather + WMMA GEMM (x@W+b) + fused LayerNorm ======== */
__global__ __launch_bounds__(256) void k_lin_ln(
    const float* features, const int* idxs,
    const float* proto_root, const float* proto_neigh,
    const float* xw, const float* xb, const float* g, const float* bb,
    float* h_all)
{
  /* W cached in LDS in fragment order: frag = 2x ds_load_b128 */
  __shared__ __align__(32) h16 Wshuf[FIN_*DX_];       /* 32 KB */
  for (int f = threadIdx.x; f < FIN_*DX_; f += 256) {
    int i = f & 15, lane = (f >> 4) & 31, tile = f >> 9;
    int nt = tile & 7, ks = tile >> 3;
    int row = ks*32 + ((lane >> 4) << 4) + i;
    int col = nt*16 + (lane & 15);
    Wshuf[f] = (h16)xw[row*DX_ + col];
  }
  __syncthreads();
  const v16h* Wv = (const v16h*)Wshuf;

  int wid   = blockIdx.x * 8 + (threadIdx.x >> 5);
  int lane  = threadIdx.x & 31;
  int rowbase = wid * 16;
  int m16 = lane & 15;
  int hh  = lane >> 4;
  int lane_hi8 = hh * 8;

  /* resolve per-lane source row (regions are 16-row aligned, never straddle) */
  int row = rowbase + m16;
  const float* src;
  if (row < BM_) {
    int idx = idxs[row];
    src = (idx == NALL_) ? nullptr : (features + (long)idx * FIN_);
  } else if (row < PN_OFF) {
    src = proto_root + (long)(row - PR_OFF) * FIN_;
  } else {
    src = proto_neigh + (long)(row - PN_OFF) * FIN_;
  }

  v8f zero = {};
  v8f acc[8];
#pragma unroll
  for (int nt = 0; nt < 8; ++nt) acc[nt] = zero;

#pragma unroll
  for (int ks = 0; ks < 4; ++ks) {
    v16h a;
#pragma unroll
    for (int i = 0; i < 16; ++i) a[i] = (h16)0.f;
    if (src) a = frag_a_f32cvt(src + ks*32, lane_hi8);
#pragma unroll
    for (int nt = 0; nt < 8; ++nt) {
      v16h bf = Wv[(ks*8 + nt)*32 + lane];
      acc[nt] = wmma16(a, bf, acc[nt]);
    }
  }

  /* bias + LayerNorm directly on C fragments; row stats via half-wave shfl */
  float s1[8], s2[8];
#pragma unroll
  for (int j = 0; j < 8; ++j) { s1[j] = 0.f; s2[j] = 0.f; }
#pragma unroll
  for (int nt = 0; nt < 8; ++nt) {
    float bn = xb[nt*16 + m16];
#pragma unroll
    for (int j = 0; j < 8; ++j) {
      float x = acc[nt][j] + bn;
      acc[nt][j] = x;
      s1[j] += x; s2[j] += x*x;
    }
  }
#pragma unroll
  for (int msk = 1; msk < 16; msk <<= 1) {
#pragma unroll
    for (int j = 0; j < 8; ++j) {
      s1[j] += __shfl_xor(s1[j], msk, 32);
      s2[j] += __shfl_xor(s2[j], msk, 32);
    }
  }
  float mean[8], inv[8];
#pragma unroll
  for (int j = 0; j < 8; ++j) {
    mean[j] = s1[j] * (1.0f/DX_);
    float var = s2[j] * (1.0f/DX_) - mean[j]*mean[j];
    inv[j] = rsqrtf(var + 1e-5f);
  }
#pragma unroll
  for (int nt = 0; nt < 8; ++nt) {
    int col = nt*16 + m16;
    float gg = g[col], bv = bb[col];
#pragma unroll
    for (int j = 0; j < 8; ++j) {
      int r = rowbase + hh*8 + j;
      h_all[(long)r * DX_ + col] = (acc[nt][j] - mean[j]) * inv[j] * gg + bv;
    }
  }
}

/* ============ k_proto: prototype-side precompute (fragment-ordered Y) ======= */
__global__ __launch_bounds__(256) void k_proto(
    const float* h_all, const float* tnx, const float* tns,
    const float* wn_w1, const float* proto_dn, const float* proto_rad,
    const float* s_g, const float* s_b,
    h16* hpshuf, float* hp_n2, float* hp2w, h16* Yf, h16* Ys, float* rps)
{
  int gid = blockIdx.x * blockDim.x + threadIdx.x;
  int nth = gridDim.x * blockDim.x;
  const float* hpr = h_all + (long)PR_OFF * DX_;
  const float* hpn = h_all + (long)PN_OFF * DX_;

  for (int k = gid; k < K_; k += nth) {   /* root proto: B-shuffled f16 + norms */
    float s = 0.f;
    for (int c = 0; c < DX_; ++c) {
      float v = hpr[k*DX_ + c];
      hpshuf[remapB(c, k, 4)] = (h16)v;   /* B element (row=dim c, col=proto k) */
      s += v*v;
    }
    hp_n2[k] = s;
    float v[NN_];
#pragma unroll
    for (int m = 0; m < NN_; ++m) v[m] = proto_rad[k*NN_ + m];
    sort9(v);
#pragma unroll
    for (int m = 0; m < NN_; ++m) rps[k*NN_ + m] = v[m];
  }

  for (int t = gid; t < K_*32; t += nth) {       /* hp @ wn_w1[DX:] */
    int k = t >> 5, j = t & 31;
    float s = 0.f;
    for (int c = 0; c < DX_; ++c) s += hpr[k*DX_ + c] * wn_w1[(DX_ + c)*32 + j];
    hp2w[t] = s;
  }

  for (int t = gid; t < K_*L_; t += nth) {       /* feature-side pps -> Yf */
    int k = t >> 5, l = t & 31;
    float p[NN_];
#pragma unroll
    for (int m = 0; m < NN_; ++m) {
      const float* hr = hpn + (long)(k*NN_ + m) * DX_;
      const float* tr = tnx + l*DX_;
      float s = 0.f;
      for (int c = 0; c < DX_; ++c) s += hr[c]*tr[c];
      p[m] = s;
    }
    sort9(p);
#pragma unroll
    for (int m = 0; m < NN_; ++m) {
      Yf[remapB(m*L_ + l, k, 4)]          = (h16)(-2.0f/L_ * p[m]);
      Yf[remapB(NN_*L_ + m*L_ + l, k, 4)] = (h16)(p[m]*p[m]*(1.0f/L_));
    }
  }

  for (int k = gid; k < K_; k += nth) {          /* structural protos -> Ys */
    float Cm[NN_*NN_];
#pragma unroll
    for (int i = 0; i < NN_*NN_; ++i) Cm[i] = 0.f;
    int p = 0;
#pragma unroll
    for (int i = 0; i < NN_; ++i)
#pragma unroll
      for (int j = i+1; j < NN_; ++j) {
        float v = 1.f/(1.f + expf(-proto_dn[p*K_ + k]));
        Cm[i*NN_+j] = v; Cm[j*NN_+i] = v; ++p;
      }
#pragma unroll
    for (int j = 0; j < NN_; ++j) {              /* sort each column over rows */
      float c[NN_];
#pragma unroll
      for (int i = 0; i < NN_; ++i) c[i] = Cm[i*NN_+j];
      sort9(c);
#pragma unroll
      for (int i = 0; i < NN_; ++i) Cm[i*NN_+j] = c[i];
    }
#pragma unroll
    for (int i = 0; i < NN_; ++i) {              /* LN rows (s_ln params) */
      float mu = 0.f;
#pragma unroll
      for (int j = 0; j < NN_; ++j) mu += Cm[i*NN_+j];
      mu *= (1.0f/NN_);
      float var = 0.f;
#pragma unroll
      for (int j = 0; j < NN_; ++j) { float d = Cm[i*NN_+j]-mu; var += d*d; }
      var *= (1.0f/NN_);
      float inv = rsqrtf(var + 1e-5f);
#pragma unroll
      for (int j = 0; j < NN_; ++j)
        Cm[i*NN_+j] = (Cm[i*NN_+j]-mu)*inv*s_g[j] + s_b[j];
    }
    for (int l = 0; l < L_; ++l) {
      float pv[NN_];
#pragma unroll
      for (int m = 0; m < NN_; ++m) {
        float s = 0.f;
#pragma unroll
        for (int d = 0; d < NN_; ++d) s += Cm[m*NN_+d]*tns[l*NN_+d];
        pv[m] = s;
      }
      sort9(pv);
#pragma unroll
      for (int m = 0; m < NN_; ++m) {
        Ys[remapB(m*L_ + l, k, 4)]          = (h16)(-2.0f/L_ * pv[m]);
        Ys[remapB(NN_*L_ + m*L_ + l, k, 4)] = (h16)(pv[m]*pv[m]*(1.0f/L_));
      }
    }
  }
}

/* ============ k_graph: BFS dists + structural sw prep + radial (1 wave / b) == */
__global__ __launch_bounds__(256) void k_graph(
    const float* adj, const int* idxs, const float* tns,
    const float* s_g, const float* s_b,
    h16* Xs, float* t1ms, float* c0rad, float* wrr, float* wrv)
{
  __shared__ float g_adj[8][100], g_c0[8][100], g_c1[8][100], g_d[8][100];
  __shared__ float g_fm[8][16], g_hs[8][81];
  int w = threadIdx.x >> 5, lane = threadIdx.x & 31;
  int b = blockIdx.x * 8 + w;
  float* A  = g_adj[w]; float* C0 = g_c0[w]; float* C1 = g_c1[w];
  float* D  = g_d[w];   float* FM = g_fm[w]; float* HS = g_hs[w];

  if (lane < 10)
    FM[lane] = (lane == 0) ? 1.f : ((idxs[b*M_ + lane] != NALL_) ? 1.f : 0.f);
  for (int e = lane; e < 100; e += 32) {
    int i = e / 10, j = e % 10;
    float a = (adj[(long)b*100 + e] > 1e-5f) ? 1.f : 0.f;
    A[e] = a; C0[e] = a;
    D[e] = (i == j) ? 0.f : ((a > 0.f) ? 1.f : 10.f);
  }
  __syncthreads();
  for (int k = 2; k <= 9; ++k) {                 /* boolean matrix powers */
    float* src = (k & 1) ? C1 : C0;
    float* dst = (k & 1) ? C0 : C1;
    for (int e = lane; e < 100; e += 32) {
      int i = e / 10, j = e % 10;
      float s = 0.f;
      for (int t = 0; t < 10; ++t) s += src[i*10+t]*A[t*10+j];
      dst[e] = s;
      if (s > 0.f && D[e] == 10.f) D[e] = (float)k;
    }
    __syncthreads();
  }
  for (int e = lane; e < 100; e += 32) {
    int i = e / 10, j = e % 10;
    float d = (FM[i]*FM[j] == 0.f) ? 10.f : D[e];
    D[e] = d * 0.1f;
  }
  __syncthreads();

  /* radial prep (all lanes compute redundantly; lane 0 stores) */
  {
    float v[9], wv[9], cnt = 0.f;
#pragma unroll
    for (int m = 0; m < 9; ++m) { v[m] = D[m+1]; wv[m] = FM[m+1]; cnt += wv[m]; }
    sort9p(v, wv);
    float inv = 1.f/(cnt + 1e-9f), c0 = 0.f;
#pragma unroll
    for (int m = 0; m < 9; ++m) { wv[m] *= inv; c0 += wv[m]*v[m]*v[m]; }
    if (lane == 0) {
      c0rad[b] = c0;
#pragma unroll
      for (int m = 0; m < 9; ++m) { wrr[b*9+m] = wv[m]*v[m]; wrv[b*9+m] = wv[m]; }
    }
  }
  /* hs_neigh: sort columns over rows, then LN rows */
  if (lane < 9) {
    float c[9];
#pragma unroll
    for (int i = 0; i < 9; ++i) c[i] = D[(1+i)*10 + (1+lane)];
    sort9(c);
#pragma unroll
    for (int i = 0; i < 9; ++i) HS[i*9 + lane] = c[i];
  }
  __syncthreads();
  if (lane < 9) {
    float mu = 0.f;
#pragma unroll
    for (int j = 0; j < 9; ++j) mu += HS[lane*9+j];
    mu *= (1.0f/9.0f);
    float var = 0.f;
#pragma unroll
    for (int j = 0; j < 9; ++j) { float d = HS[lane*9+j]-mu; var += d*d; }
    var *= (1.0f/9.0f);
    float inv = rsqrtf(var + 1e-5f);
#pragma unroll
    for (int j = 0; j < 9; ++j)
      HS[lane*9+j] = (HS[lane*9+j]-mu)*inv*s_g[j] + s_b[j];
  }
  __syncthreads();
  /* structural projections: lane == l */
  {
    int l = lane;
    float pv[9], wv[9], cnt = 0.f;
#pragma unroll
    for (int m = 0; m < 9; ++m) { wv[m] = FM[m+1]; cnt += wv[m]; }
#pragma unroll
    for (int m = 0; m < 9; ++m) {
      float s = 0.f;
#pragma unroll
      for (int d = 0; d < 9; ++d) s += HS[m*9+d]*tns[l*9+d];
      pv[m] = s;
    }
    sort9p(pv, wv);
    float inv = 1.f/(cnt + 1e-9f), t1 = 0.f;
#pragma unroll
    for (int m = 0; m < 9; ++m) {
      float wn = wv[m]*inv;
      float a  = wn*pv[m];
      t1 += a*pv[m];
      Xs[(long)b*CSTR_ + m*L_ + l]          = (h16)a;
      Xs[(long)b*CSTR_ + NN_*L_ + m*L_ + l] = (h16)wn;
    }
#pragma unroll
    for (int msk = 16; msk >= 1; msk >>= 1) t1 += __shfl_xor(t1, msk, 32);
    if (lane == 0) t1ms[b] = t1 * (1.0f/L_);
  }
}

/* ============ k_feat: pooling/alpha/hb + feature sw prep (1 wave / b) ======= */
__global__ __launch_bounds__(256) void k_feat(
    const float* h_all, const int* idxs, const float* tnx,
    const float* an_w1, const float* an_b1, const float* an_w2,
    const float* an_b2, const float* alpha_raw,
    const float* wn_w1, const float* wn_b1,
    h16* Xf, float* t1mx, float* alphab, float* hbo, float* hn2o, h16* hrootf)
{
  __shared__ float hsh_s[8][M_*DX_];
  __shared__ float psh_s[8][DX_];
  int w = threadIdx.x >> 5, lane = threadIdx.x & 31;
  int b = blockIdx.x * 8 + w;
  float* hsh = hsh_s[w]; float* psh = psh_s[w];
  const float* hbase = h_all + (long)b * M_ * DX_;
  for (int i = lane; i < M_*DX_; i += 32) hsh[i] = hbase[i];
  float vm[9], cnt = 0.f;
#pragma unroll
  for (int m = 0; m < 9; ++m) {
    vm[m] = (idxs[b*M_ + 1 + m] != NALL_) ? 1.f : 0.f;
    cnt += vm[m];
  }
  __syncthreads();
  float invc = 1.f/(cnt + 1e-9f);
#pragma unroll
  for (int cc = 0; cc < 4; ++cc) {               /* masked mean pooling */
    int c = lane*4 + cc;
    float s = 0.f;
#pragma unroll
    for (int m = 0; m < 9; ++m) s += vm[m]*hsh[(1+m)*DX_ + c];
    psh[c] = s * invc;
  }
  __syncthreads();
  {                                              /* alpha MLP */
    int j = lane;
    float t = an_b1[j];
    for (int c = 0; c < DX_; ++c) t += psh[c]*an_w1[c*32 + j];
    t = fmaxf(t, 0.f);
    float contrib = t * an_w2[j];
#pragma unroll
    for (int msk = 16; msk >= 1; msk >>= 1) contrib += __shfl_xor(contrib, msk, 32);
    if (lane == 0) {
      float logit = alpha_raw[0] + contrib + an_b2[0];
      alphab[b] = 1.f/(1.f + expf(-logit));
    }
  }
  {                                              /* hb = h_root @ wn_w1[:DX] + b */
    int j = lane;
    float t = wn_b1[j];
    for (int c = 0; c < DX_; ++c) t += hsh[c]*wn_w1[c*32 + j];
    hbo[b*32 + j] = t;
  }
  {                                              /* ||h_root||^2 + f16 copy */
    float s = 0.f;
#pragma unroll
    for (int cc = 0; cc < 4; ++cc) {
      int c = lane*4 + cc;
      float v = hsh[c];
      s += v*v;
      hrootf[(long)b*DX_ + c] = (h16)v;
    }
#pragma unroll
    for (int msk = 16; msk >= 1; msk >>= 1) s += __shfl_xor(s, msk, 32);
    if (lane == 0) hn2o[b] = s;
  }
  {                                              /* feature projections: lane==l */
    int l = lane;
    float pv[9], wv[9];
#pragma unroll
    for (int m = 0; m < 9; ++m) {
      const float* hr = hsh + (1+m)*DX_;
      const float* tr = tnx + l*DX_;
      float s = 0.f;
      for (int c = 0; c < DX_; ++c) s += hr[c]*tr[c];
      pv[m] = s; wv[m] = vm[m];
    }
    sort9p(pv, wv);
    float t1 = 0.f;
#pragma unroll
    for (int m = 0; m < 9; ++m) {
      float wn = wv[m]*invc;
      float a  = wn*pv[m];
      t1 += a*pv[m];
      Xf[(long)b*CSTR_ + m*L_ + l]          = (h16)a;
      Xf[(long)b*CSTR_ + NN_*L_ + m*L_ + l] = (h16)wn;
    }
#pragma unroll
    for (int msk = 16; msk >= 1; msk >>= 1) t1 += __shfl_xor(t1, msk, 32);
    if (lane == 0) t1mx[b] = t1 * (1.0f/L_);
  }
}

/* ============ k_final: 3 WMMA GEMMs per 16x64 tile + gated combine ========== */
__global__ __launch_bounds__(256) void k_final(
    const h16* Xs, const h16* Ys, const h16* Xf, const h16* Yf,
    const h16* hrootf, const h16* hpshuf,
    const float* t1ms, const float* t1mx, const float* hn2, const float* hp_n2,
    const float* c0rad, const float* wrr, const float* wrv, const float* rps,
    const float* alphab, const float* hb, const float* hp2w,
    const float* wn_w2, const float* wn_b2, const float* w_raw,
    const float* log_gamma, float* out)
{
  int wid  = blockIdx.x * 8 + (threadIdx.x >> 5);
  int lane = threadIdx.x & 31;
  int rowbase = wid * 16;
  int m16 = lane & 15, hh = lane >> 4;
  int lane_hi8 = hh*8;

  v8f zero = {};
  v8f aS[4], aF[4], aR[4];
#pragma unroll
  for (int nt = 0; nt < 4; ++nt) { aS[nt]=zero; aF[nt]=zero; aR[nt]=zero; }

  const h16* xsr = Xs + (long)(rowbase + m16)*CSTR_;
  const h16* xfr = Xf + (long)(rowbase + m16)*CSTR_;
  const h16* xrr = hrootf + (long)(rowbase + m16)*DX_;
  const v16h* Ysv = (const v16h*)Ys;
  const v16h* Yfv = (const v16h*)Yf;
  const v16h* Hpv = (const v16h*)hpshuf;
  __builtin_prefetch(xsr, 0, 1);
  __builtin_prefetch(xfr, 0, 1);

  for (int ks = 0; ks < 18; ++ks) {              /* structural SW GEMM */
    v16h a = frag_a_f16(xsr + ks*32, lane_hi8);
#pragma unroll
    for (int nt = 0; nt < 4; ++nt)
      aS[nt] = wmma16(a, Ysv[(ks*4 + nt)*32 + lane], aS[nt]);
  }
  for (int ks = 0; ks < 18; ++ks) {              /* feature SW GEMM */
    v16h a = frag_a_f16(xfr + ks*32, lane_hi8);
#pragma unroll
    for (int nt = 0; nt < 4; ++nt)
      aF[nt] = wmma16(a, Yfv[(ks*4 + nt)*32 + lane], aF[nt]);
  }
#pragma unroll
  for (int ks = 0; ks < 4; ++ks) {               /* h_root . h_proto_root GEMM */
    v16h a = frag_a_f16(xrr + ks*32, lane_hi8);
#pragma unroll
    for (int nt = 0; nt < 4; ++nt)
      aR[nt] = wmma16(a, Hpv[(ks*4 + nt)*32 + lane], aR[nt]);
  }

  float gamma = expf(log_gamma[0]);
  float wraw = w_raw[0], wb2 = wn_b2[0];

#pragma unroll
  for (int j = 0; j < 8; ++j) {
    int br = rowbase + hh*8 + j;
    float t1s = t1ms[br], t1x = t1mx[br], n2b = hn2[br];
    float c0r = c0rad[br], alph = alphab[br];
    const float* hbr  = hb  + br*32;
    const float* wrrr = wrr + br*9;
    const float* wrvr = wrv + br*9;
#pragma unroll
    for (int nt = 0; nt < 4; ++nt) {
      int kc = nt*16 + m16;
      float S = aS[nt][j] + t1s;
      float F = aF[nt][j] + t1x;
      float Droot = n2b + hp_n2[kc] - 2.f*aR[nt][j];
      float rad = c0r;
      const float* rr = rps + kc*9;
      for (int m = 0; m < 9; ++m) {
        float r = rr[m];
        rad += wrvr[m]*r*r - 2.f*wrrr[m]*r;
      }
      float logit = wb2;
      const float* hpw = hp2w + kc*32;
      for (int j2 = 0; j2 < 32; ++j2)
        logit += fmaxf(hbr[j2] + hpw[j2], 0.f) * wn_w2[j2];
      float wgt = 1.f/(1.f + expf(-(wraw + logit)));
      float d_feat = wgt*Droot + (1.f-wgt)*F;
      float d_str  = wgt*rad   + (1.f-wgt)*S;
      out[(long)br*K_ + kc] = expf(-gamma*(alph*d_feat + (1.f-alph)*d_str));
    }
  }
}

/* ============================ host launcher ================================= */
extern "C" void kernel_launch(void* const* d_in, const int* in_sizes, int n_in,
                              void* d_out, int out_size, void* d_ws, size_t ws_size,
                              hipStream_t stream) {
  const float* adj        = (const float*)d_in[0];
  const float* features   = (const float*)d_in[1];
  const int*   idxs       = (const int*)d_in[2];
  const float* x_lin_w    = (const float*)d_in[3];
  const float* x_lin_b    = (const float*)d_in[4];
  const float* x_ln_g     = (const float*)d_in[5];
  const float* x_ln_b     = (const float*)d_in[6];
  const float* s_ln_g     = (const float*)d_in[7];
  const float* s_ln_b     = (const float*)d_in[8];
  const float* theta_x    = (const float*)d_in[9];
  const float* theta_s    = (const float*)d_in[10];
  const float* alpha_raw  = (const float*)d_in[11];
  const float* an_w1      = (const float*)d_in[12];
  const float* an_b1      = (const float*)d_in[13];
  const float* an_w2      = (const float*)d_in[14];
  const float* an_b2      = (const float*)d_in[15];
  const float* wn_w1      = (const float*)d_in[16];
  const float* wn_b1      = (const float*)d_in[17];
  const float* wn_w2      = (const float*)d_in[18];
  const float* wn_b2      = (const float*)d_in[19];
  const float* w_raw      = (const float*)d_in[20];
  const float* proto_root  = (const float*)d_in[21];
  const float* proto_neigh = (const float*)d_in[22];
  const float* proto_rad   = (const float*)d_in[23];
  const float* proto_dn    = (const float*)d_in[24];
  const float* log_gamma   = (const float*)d_in[25];
  float* out = (float*)d_out;

  char* ws = (char*)d_ws;
  size_t off = 0;
  auto alloc = [&](size_t bytes) -> char* {
    char* p = ws + off;
    off = (off + bytes + 255) & ~(size_t)255;
    return p;
  };

  float* h_all  = (float*)alloc((size_t)RTOT_*DX_*4);
  float* tnx    = (float*)alloc(L_*DX_*4);
  float* tns    = (float*)alloc(L_*NN_*4);
  h16*   hpf    = (h16*)  alloc(K_*DX_*2);
  float* hp_n2  = (float*)alloc(K_*4);
  float* hp2w   = (float*)alloc(K_*32*4);
  float* rps    = (float*)alloc(K_*NN_*4);
  h16*   Ys     = (h16*)  alloc((size_t)CSTR_*K_*2);
  h16*   Yf     = (h16*)  alloc((size_t)CSTR_*K_*2);
  h16*   Xs     = (h16*)  alloc((size_t)B_*CSTR_*2);
  h16*   Xf     = (h16*)  alloc((size_t)B_*CSTR_*2);
  h16*   hrootf = (h16*)  alloc((size_t)B_*DX_*2);
  float* t1ms   = (float*)alloc(B_*4);
  float* t1mx   = (float*)alloc(B_*4);
  float* c0radv = (float*)alloc(B_*4);
  float* wrrv   = (float*)alloc(B_*NN_*4);
  float* wrvv   = (float*)alloc(B_*NN_*4);
  float* alphab = (float*)alloc(B_*4);
  float* hbv    = (float*)alloc(B_*32*4);
  float* hn2v   = (float*)alloc(B_*4);

  k_prep<<<dim3(1), dim3(32), 0, stream>>>(theta_x, theta_s, tnx, tns);
  k_lin_ln<<<dim3(RTOT_/16/8), dim3(256), 0, stream>>>(
      features, idxs, proto_root, proto_neigh,
      x_lin_w, x_lin_b, x_ln_g, x_ln_b, h_all);
  k_proto<<<dim3(16), dim3(256), 0, stream>>>(
      h_all, tnx, tns, wn_w1, proto_dn, proto_rad, s_ln_g, s_ln_b,
      hpf, hp_n2, hp2w, Yf, Ys, rps);
  k_graph<<<dim3(B_/8), dim3(256), 0, stream>>>(
      adj, idxs, tns, s_ln_g, s_ln_b, Xs, t1ms, c0radv, wrrv, wrvv);
  k_feat<<<dim3(B_/8), dim3(256), 0, stream>>>(
      h_all, idxs, tnx, an_w1, an_b1, an_w2, an_b2, alpha_raw, wn_w1, wn_b1,
      Xf, t1mx, alphab, hbv, hn2v, hrootf);
  k_final<<<dim3(B_/16/8), dim3(256), 0, stream>>>(
      Xs, Ys, Xf, Yf, hrootf, hpf, t1ms, t1mx, hn2v, hp_n2,
      c0radv, wrrv, wrvv, rps, alphab, hbv, hp2w,
      wn_w2, wn_b2, w_raw, log_gamma, out);

  (void)n_in; (void)in_sizes; (void)out_size; (void)ws_size;
}
